// DropoutTransformer_88476326298320
// MI455X (gfx1250) — matrix operand order repaired and checked by reading, not compiled, not run on hardware
//
#include <hip/hip_runtime.h>

// ---------------------------------------------------------------------------
// Model dimensions (fixed by the reference)
// ---------------------------------------------------------------------------
#define NBATCH 2
#define NT   1024
#define ND   1024
#define NH   16
#define NHS  64
#define NL   4
#define NV   32000
#define NM   (NBATCH * NT)          // 2048 rows

typedef unsigned short ushort_t;
typedef __attribute__((ext_vector_type(16))) __bf16        v16bf;
typedef __attribute__((ext_vector_type(8)))  float         v8f;
typedef __attribute__((ext_vector_type(8)))  unsigned int  u32x8;

// ---------------------------------------------------------------------------
// Helpers
// ---------------------------------------------------------------------------
__device__ __forceinline__ ushort_t f2bf(float f) {
    unsigned u = __builtin_bit_cast(unsigned, f);
    unsigned r = (u + 0x7FFFu + ((u >> 16) & 1u)) >> 16;   // RNE
    return (ushort_t)r;
}
__device__ __forceinline__ unsigned pack2bf(float a, float b) {
    return (unsigned)f2bf(a) | ((unsigned)f2bf(b) << 16);
}
__device__ __forceinline__ v8f v8f_zero() {
    v8f r;
    #pragma unroll
    for (int i = 0; i < 8; ++i) r[i] = 0.0f;
    return r;
}

// D = A*B + C  (16x16x32 bf16 -> f32 accumulate)
__device__ __forceinline__ v8f wmma_bf16(v16bf a, v16bf b, v8f c) {
    return __builtin_amdgcn_wmma_f32_16x16x32_bf16(
        false, a, false, b, (short)0, c, false, false);
}

// CDNA5 async global->LDS copy of 16 bytes per lane (ASYNCcnt-tracked).
// LDS address = low 32 bits of the generic pointer (flat->LDS uses addr[31:0]).
__device__ __forceinline__ void async_copy16(const void* g, void* lds) {
    const unsigned l = (unsigned)(size_t)lds;
    asm volatile("global_load_async_to_lds_b128 %0, %1, off"
                 :: "v"(l), "v"(g) : "memory");
}
template <int N>
__device__ __forceinline__ void async_wait_le() {
    asm volatile("s_wait_asynccnt %0" :: "i"(N) : "memory");
}

// A fragment (16x32 bf16) from LDS tile stored [m][k] (k-length = stride_s).
__device__ __forceinline__ v16bf frag_a_lds(const ushort_t* tile, int stride_s, int lane) {
    const int hl = lane >> 4, lr = lane & 15;
    const unsigned* p = (const unsigned*)(tile + lr * stride_s);
    u32x8 r;
    #pragma unroll
    for (int v = 0; v < 8; ++v) {
        const int idx = v + 8 * hl;   // = kk/2
        r[v] = p[idx];
    }
    return __builtin_bit_cast(v16bf, r);
}

// B fragment (32x16 bf16) from LDS tile stored [n][k] (k contiguous).
__device__ __forceinline__ v16bf frag_b_lds(const ushort_t* tile, int stride_s, int lane) {
    const int hl = lane >> 4, lc = lane & 15;
    const unsigned* p = (const unsigned*)(tile + lc * stride_s);
    u32x8 r;
    #pragma unroll
    for (int j = 0; j < 8; ++j) r[j] = p[8 * hl + j];
    return __builtin_bit_cast(v16bf, r);
}

// ---------------------------------------------------------------------------
// Embedding: h[m][d] = tok_emb[x[m]][d] + pos_emb[m % T][d]  (fp32 residual)
// ---------------------------------------------------------------------------
__global__ __launch_bounds__(256)
void embed_k(const int* __restrict__ x, const float* __restrict__ te,
             const float* __restrict__ pe, float* __restrict__ h) {
    const int gid = blockIdx.x * 256 + threadIdx.x;
    const int m = gid >> 8;
    const int d = (gid & 255) << 2;
    const int t = m & (NT - 1);
    const int tok = x[m];
    const float4 a = *(const float4*)(te + (size_t)tok * ND + d);
    const float4 p = *(const float4*)(pe + (size_t)t * ND + d);
    float4 o; o.x = a.x + p.x; o.y = a.y + p.y; o.z = a.z + p.z; o.w = a.w + p.w;
    *(float4*)(h + (size_t)m * ND + d) = o;
}

// ---------------------------------------------------------------------------
// LayerNorm over D=1024, fp32 in -> bf16 out. One block per row.
// ---------------------------------------------------------------------------
__global__ __launch_bounds__(256)
void layernorm_k(const float* __restrict__ x, const float* __restrict__ g,
                 const float* __restrict__ b, ushort_t* __restrict__ y) {
    __shared__ float red[256];
    const int row = blockIdx.x;
    const int tid = threadIdx.x;
    const float4 v = *(const float4*)(x + (size_t)row * ND + tid * 4);
    red[tid] = v.x + v.y + v.z + v.w;
    __syncthreads();
    #pragma unroll
    for (int o = 128; o > 0; o >>= 1) {
        if (tid < o) red[tid] += red[tid + o];
        __syncthreads();
    }
    const float mu = red[0] * (1.0f / ND);
    __syncthreads();
    float4 c; c.x = v.x - mu; c.y = v.y - mu; c.z = v.z - mu; c.w = v.w - mu;
    red[tid] = c.x * c.x + c.y * c.y + c.z * c.z + c.w * c.w;
    __syncthreads();
    #pragma unroll
    for (int o = 128; o > 0; o >>= 1) {
        if (tid < o) red[tid] += red[tid + o];
        __syncthreads();
    }
    const float rs = rsqrtf(red[0] * (1.0f / ND) + 1e-5f);
    const int i = tid * 4;
    uint2 u;
    u.x = pack2bf(c.x * rs * g[i + 0] + b[i + 0], c.y * rs * g[i + 1] + b[i + 1]);
    u.y = pack2bf(c.z * rs * g[i + 2] + b[i + 2], c.w * rs * g[i + 3] + b[i + 3]);
    *(uint2*)(y + (size_t)row * ND + i) = u;
}

// ---------------------------------------------------------------------------
// One-time weight conversion fp32 -> bf16 with transpose to [N][K]
// (k contiguous), so GEMM B-tiles become raw async byte copies.
// Source element (k, n) at  W + (n/nb)*K*nb + k*nb + (n%nb).
// Grid: (N/32, K/32), 256 threads, LDS-tiled transpose.
// ---------------------------------------------------------------------------
__global__ __launch_bounds__(256)
void convert_tr_k(const float* __restrict__ W, ushort_t* __restrict__ Wt,
                  int K, int nb) {
    __shared__ ushort_t t[32][33];
    const int n0 = blockIdx.x * 32;
    const int k0 = blockIdx.y * 32;
    const int tx = threadIdx.x & 31;
    const int ty = threadIdx.x >> 5;
    const int n = n0 + tx;
    const float* src = W + (size_t)(n / nb) * K * nb + (n % nb);
    #pragma unroll
    for (int i = 0; i < 4; ++i) {
        const int k = k0 + ty * 4 + i;
        t[tx][ty * 4 + i] = f2bf(src[(size_t)k * nb]);   // coalesced along n
    }
    __syncthreads();
    #pragma unroll
    for (int i = 0; i < 4; ++i) {
        const int nr = ty * 4 + i;
        Wt[(size_t)(n0 + nr) * K + k0 + tx] = t[nr][tx];  // coalesced along k
    }
}

// ---------------------------------------------------------------------------
// WMMA GEMM, all-bf16 operands, async double-buffered LDS staging.
//   A  : bf16 [M][K] row-major
//   Wt : bf16 [N][K] (pre-transposed weights)
// Block tile 128x128, K-step 32; 8 waves, each a 32x64 tile (2x4 WMMA).
// Epilogues: 0 bf16 store [m][n]; 1 bias+relu -> bf16 [m][n];
//            2 bias+ldrop, fp32 residual += ; 3 bias -> fp32 (logits);
//            4 bf16 store transposed [n][m] (vectorized, for V).
// ---------------------------------------------------------------------------
template <int EP>
__global__ __launch_bounds__(256)
void gemm_wmma(const ushort_t* __restrict__ A, const ushort_t* __restrict__ Wt,
               void* __restrict__ Cv, const float* __restrict__ bias,
               const float* __restrict__ dA, const float* __restrict__ dB,
               int M, int N, int K) {
    __shared__ ushort_t lsA[2][128 * 32];
    __shared__ ushort_t lsB[2][128 * 32];

    const int tid  = threadIdx.x;
    const int lane = tid & 31;
    const int wave = tid >> 5;
    const int wm   = wave & 3;
    const int wn   = wave >> 2;
    const int m0   = blockIdx.y * 128;
    const int n0   = blockIdx.x * 128;

    // async-stage one 128x32 A tile + 128x32 B tile (4 ops per wave)
    auto stage = [&](int kt) {
        const int k0  = kt << 5;
        const int buf = kt & 1;
        #pragma unroll
        for (int it = 0; it < 2; ++it) {
            const int idx = tid + it * 256;
            const int r   = idx >> 2;          // 0..127
            const int c   = (idx & 3) << 3;    // bf16 offset, 16B chunks
            async_copy16(A  + (size_t)(m0 + r) * K + k0 + c, &lsA[buf][r * 32 + c]);
            async_copy16(Wt + (size_t)(n0 + r) * K + k0 + c, &lsB[buf][r * 32 + c]);
        }
    };

    v8f acc[2][4];
    #pragma unroll
    for (int i = 0; i < 2; ++i)
        #pragma unroll
        for (int j = 0; j < 4; ++j) acc[i][j] = v8f_zero();

    const int ktiles = K >> 5;
    stage(0);
    for (int kt = 0; kt < ktiles; ++kt) {
        if (kt + 1 < ktiles) { stage(kt + 1); async_wait_le<4>(); }
        else                 {                async_wait_le<0>(); }
        __syncthreads();
        const int buf = kt & 1;
        v16bf af[2], wf[4];
        #pragma unroll
        for (int i = 0; i < 2; ++i)
            af[i] = frag_a_lds(&lsA[buf][(wm * 32 + i * 16) * 32], 32, lane);
        #pragma unroll
        for (int j = 0; j < 4; ++j)
            wf[j] = frag_b_lds(&lsB[buf][(wn * 64 + j * 16) * 32], 32, lane);
        #pragma unroll
        for (int i = 0; i < 2; ++i)
            #pragma unroll
            for (int j = 0; j < 4; ++j)
                acc[i][j] = wmma_bf16(af[i], wf[j], acc[i][j]);
        __syncthreads();   // protect buffer reused by stage(kt+2)
    }

    const int hl = lane >> 4, lc = lane & 15;
    if constexpr (EP == 4) {
        // transposed bf16 store: rows along m are contiguous -> uint4
        #pragma unroll
        for (int i = 0; i < 2; ++i) {
            const int mm0 = m0 + wm * 32 + i * 16 + 8 * hl;
            #pragma unroll
            for (int j = 0; j < 4; ++j) {
                const int nn = n0 + wn * 64 + j * 16 + lc;
                uint4 u;
                u.x = pack2bf(acc[i][j][0], acc[i][j][1]);
                u.y = pack2bf(acc[i][j][2], acc[i][j][3]);
                u.z = pack2bf(acc[i][j][4], acc[i][j][5]);
                u.w = pack2bf(acc[i][j][6], acc[i][j][7]);
                *(uint4*)((ushort_t*)Cv + (size_t)nn * M + mm0) = u;
            }
        }
    } else {
        #pragma unroll
        for (int i = 0; i < 2; ++i) {
            #pragma unroll
            for (int j = 0; j < 4; ++j) {
                const int nn = n0 + wn * 64 + j * 16 + lc;
                #pragma unroll
                for (int r = 0; r < 8; ++r) {
                    const int mm = m0 + wm * 32 + i * 16 + r + 8 * hl;
                    float v = acc[i][j][r];
                    if constexpr (EP == 0) {
                        ((ushort_t*)Cv)[(size_t)mm * N + nn] = f2bf(v);
                    } else if constexpr (EP == 1) {
                        v += bias[nn];
                        ((ushort_t*)Cv)[(size_t)mm * N + nn] = f2bf(v > 0.0f ? v : 0.0f);
                    } else if constexpr (EP == 2) {
                        v += bias[nn];
                        const float d = v * (0.5f * __cosf(dA[nn] * v + dB[nn]) + 0.5f);
                        ((float*)Cv)[(size_t)mm * N + nn] += d;   // residual
                    } else {
                        ((float*)Cv)[(size_t)mm * N + nn] = v + bias[nn];
                    }
                }
            }
        }
    }
}

// ---------------------------------------------------------------------------
// Fused causal attention, softmax + learned cosine dropout on probs.
//   Q, K : bf16 [m = b*T+t][h*HS+s]      (K staged async to LDS, [key][hs])
//   vT   : bf16 [n = h*HS+s][m]          (P@V B-frags read直 from global)
//   O    : bf16 [m][h*HS+s]
// Two passes over 32-key blocks (ldrop is nonlinear in normalized probs).
// ---------------------------------------------------------------------------
__global__ __launch_bounds__(256)
void attention_k(const ushort_t* __restrict__ Q, const ushort_t* __restrict__ Kb,
                 const ushort_t* __restrict__ vT, const float* __restrict__ a1,
                 const float* __restrict__ b1, ushort_t* __restrict__ O) {
    __shared__ ushort_t lsK[2][32 * 64];     // [key][hs], double-buffered
    __shared__ ushort_t lsP[8][16 * 32];     // per-wave probs [m][u]

    const int tid  = threadIdx.x;
    const int lane = tid & 31;
    const int wave = tid >> 5;
    const int hl   = lane >> 4, lc = lane & 15;
    const int q0   = blockIdx.x * 128;
    const int hh   = blockIdx.y;
    const int bb   = blockIdx.z;
    const int row0 = q0 + wave * 16;
    const size_t base = ((size_t)bb * NT) * ND + (size_t)hh * NHS;
    const float scale = 0.125f;              // HS^-0.5

    auto stageK = [&](int j) {               // 1 async op per wave
        const int key = tid >> 3;
        const int c   = (tid & 7) << 3;
        async_copy16(Kb + base + (size_t)(j * 32 + key) * ND + c,
                     &lsK[j & 1][key * 64 + c]);
    };

    // Q fragments (16x64 -> two K=32 chunks), register-resident
    v16bf qf[2];
    {
        const ushort_t* qrow = Q + base + (size_t)(row0 + lc) * ND;
        #pragma unroll
        for (int kc = 0; kc < 2; ++kc) {
            u32x8 r;
            #pragma unroll
            for (int v = 0; v < 8; ++v) {
                const int kk = kc * 32 + 2 * v + 16 * hl;
                r[v] = *(const unsigned*)(qrow + kk);
            }
            qf[kc] = __builtin_bit_cast(v16bf, r);
        }
    }

    float rm[8], rl[8];
    #pragma unroll
    for (int r = 0; r < 8; ++r) { rm[r] = -1e30f; rl[r] = 0.0f; }

    const int nkb   = (q0 + 128) >> 5;
    const int wqmax = row0 + 15;

    // ---------------- pass 1: online row max / sum ----------------
    stageK(0);
    for (int j = 0; j < nkb; ++j) {
        if (j + 1 < nkb) { stageK(j + 1); async_wait_le<1>(); }
        else             {                async_wait_le<0>(); }
        __syncthreads();
        if (j * 32 <= wqmax) {
            const ushort_t* kbuf = lsK[j & 1];
            v8f s[2];
            #pragma unroll
            for (int nt = 0; nt < 2; ++nt) {
                v8f a = v8f_zero();
                #pragma unroll
                for (int kc = 0; kc < 2; ++kc)
                    a = wmma_bf16(qf[kc], frag_b_lds(kbuf + (nt * 16) * 64 + kc * 32, 64, lane), a);
                s[nt] = a;
            }
            #pragma unroll
            for (int r = 0; r < 8; ++r) {
                const int mrow = row0 + r + 8 * hl;
                float s0 = s[0][r] * scale;
                float s1 = s[1][r] * scale;
                if (j * 32 + lc      > mrow) s0 = -1e30f;
                if (j * 32 + 16 + lc > mrow) s1 = -1e30f;
                float mx = fmaxf(s0, s1);
                #pragma unroll
                for (int o = 1; o < 16; o <<= 1) mx = fmaxf(mx, __shfl_xor(mx, o, 32));
                const float nm = fmaxf(rm[r], mx);
                float es = __expf(s0 - nm) + __expf(s1 - nm);
                #pragma unroll
                for (int o = 1; o < 16; o <<= 1) es += __shfl_xor(es, o, 32);
                rl[r] = rl[r] * __expf(rm[r] - nm) + es;
                rm[r] = nm;
            }
        }
        __syncthreads();
    }

    float il[8];
    #pragma unroll
    for (int r = 0; r < 8; ++r) il[r] = 1.0f / rl[r];

    // ---------------- pass 2: P = ldrop(softmax), O += P @ V ----------------
    v8f ao[4];
    #pragma unroll
    for (int n2 = 0; n2 < 4; ++n2) ao[n2] = v8f_zero();

    stageK(0);
    for (int j = 0; j < nkb; ++j) {
        if (j + 1 < nkb) { stageK(j + 1); async_wait_le<1>(); }
        else             {                async_wait_le<0>(); }
        __syncthreads();
        if (j * 32 <= wqmax) {
            const ushort_t* kbuf = lsK[j & 1];
            v8f s[2];
            #pragma unroll
            for (int nt = 0; nt < 2; ++nt) {
                v8f a = v8f_zero();
                #pragma unroll
                for (int kc = 0; kc < 2; ++kc)
                    a = wmma_bf16(qf[kc], frag_b_lds(kbuf + (nt * 16) * 64 + kc * 32, 64, lane), a);
                s[nt] = a;
            }
            #pragma unroll
            for (int nt = 0; nt < 2; ++nt) {
                #pragma unroll
                for (int r = 0; r < 8; ++r) {
                    const int mrow = row0 + r + 8 * hl;
                    const int u = j * 32 + nt * 16 + lc;
                    float sv = s[nt][r] * scale;
                    if (u > mrow) sv = -1e30f;
                    const float p  = __expf(sv - rm[r]) * il[r];
                    const float pd = p * (0.5f * __cosf(a1[u] * p + b1[u]) + 0.5f);
                    lsP[wave][(r + 8 * hl) * 32 + nt * 16 + lc] = f2bf(pd);
                }
            }
            const v16bf pf = frag_a_lds(lsP[wave], 32, lane);  // wave-local, dscnt-ordered
            #pragma unroll
            for (int n2 = 0; n2 < 4; ++n2) {
                // V B-fragment straight from global vT[n][m]: contiguous dwords
                const ushort_t* vrow = vT + (size_t)(hh * NHS + n2 * 16 + lc) * NM
                                          + (size_t)bb * NT + j * 32 + 16 * hl;
                u32x8 rv;
                #pragma unroll
                for (int jj = 0; jj < 8; ++jj) rv[jj] = *(const unsigned*)(vrow + 2 * jj);
                ao[n2] = wmma_bf16(pf, __builtin_bit_cast(v16bf, rv), ao[n2]);
            }
        }
        __syncthreads();
    }

    #pragma unroll
    for (int n2 = 0; n2 < 4; ++n2)
        #pragma unroll
        for (int r = 0; r < 8; ++r) {
            const int mrow = row0 + r + 8 * hl;
            O[base + (size_t)mrow * ND + n2 * 16 + lc] = f2bf(ao[n2][r]);
        }
}

// ---------------------------------------------------------------------------
// Host orchestration
// ---------------------------------------------------------------------------
extern "C" void kernel_launch(void* const* d_in, const int* in_sizes, int n_in,
                              void* d_out, int out_size, void* d_ws, size_t ws_size,
                              hipStream_t stream) {
    (void)in_sizes; (void)n_in; (void)out_size; (void)ws_size;
    const int*   x       = (const int*)  d_in[0];
    const float* tok_emb = (const float*)d_in[1];
    const float* pos_emb = (const float*)d_in[2];
    const float* qw      = (const float*)d_in[3];
    const float* kw      = (const float*)d_in[4];
    const float* vw      = (const float*)d_in[5];
    const float* a_attn1 = (const float*)d_in[6];
    const float* b_attn1 = (const float*)d_in[7];
    const float* proj_w  = (const float*)d_in[8];
    const float* proj_b  = (const float*)d_in[9];
    const float* a_attn2 = (const float*)d_in[10];
    const float* b_attn2 = (const float*)d_in[11];
    const float* ln1_g   = (const float*)d_in[12];
    const float* ln1_b   = (const float*)d_in[13];
    const float* ln2_g   = (const float*)d_in[14];
    const float* ln2_b   = (const float*)d_in[15];
    const float* ff_w1   = (const float*)d_in[16];
    const float* ff_b1   = (const float*)d_in[17];
    const float* ff_w2   = (const float*)d_in[18];
    const float* ff_b2   = (const float*)d_in[19];
    const float* a_ff    = (const float*)d_in[20];
    const float* b_ff    = (const float*)d_in[21];
    const float* lnf_g   = (const float*)d_in[22];
    const float* lnf_b   = (const float*)d_in[23];
    const float* out_w   = (const float*)d_in[24];
    const float* out_b   = (const float*)d_in[25];

    // workspace layout (byte offsets, MB-aligned)
    char* w = (char*)d_ws;
    const size_t MB = 1024 * 1024;
    ushort_t* Wq = (ushort_t*)(w + 0 * MB);     // [L*H*64][1024]   8 MB
    ushort_t* Wk = (ushort_t*)(w + 8 * MB);     //                  8 MB
    ushort_t* Wv = (ushort_t*)(w + 16 * MB);    //                  8 MB
    ushort_t* Wp = (ushort_t*)(w + 24 * MB);    // [L*1024][1024]   8 MB
    ushort_t* W1 = (ushort_t*)(w + 32 * MB);    // [L*4096][1024]  32 MB
    ushort_t* W2 = (ushort_t*)(w + 64 * MB);    // [L*1024][4096]  32 MB
    ushort_t* Wo = (ushort_t*)(w + 96 * MB);    // [32000][1024] 62.5 MB
    float*    h  = (float*)   (w + 160 * MB);   // [2048][1024] fp32
    ushort_t* xn = (ushort_t*)(w + 168 * MB);   // [2048][1024] bf16
    ushort_t* q  = (ushort_t*)(w + 172 * MB);
    ushort_t* k  = (ushort_t*)(w + 176 * MB);
    ushort_t* vT = (ushort_t*)(w + 180 * MB);   // [1024][2048] bf16
    ushort_t* ao = (ushort_t*)(w + 184 * MB);
    ushort_t* f1 = (ushort_t*)(w + 188 * MB);   // [2048][4096] bf16

    const dim3 blk(256);
    const dim3 gD(ND / 128, NM / 128);          // N=1024 GEMMs
    const dim3 gF(4 * ND / 128, NM / 128);      // N=4096 GEMM
    const dim3 gV(NV / 128, NM / 128);          // logits GEMM
    const dim3 gA(NT / 128, NH, NBATCH);        // attention

    // one-time weight convert + transpose (fp32 -> bf16 [N][K])
    convert_tr_k<<<dim3(NL * ND / 32, ND / 32), blk, 0, stream>>>(qw, Wq, ND, NHS);
    convert_tr_k<<<dim3(NL * ND / 32, ND / 32), blk, 0, stream>>>(kw, Wk, ND, NHS);
    convert_tr_k<<<dim3(NL * ND / 32, ND / 32), blk, 0, stream>>>(vw, Wv, ND, NHS);
    convert_tr_k<<<dim3(NL * ND / 32, ND / 32), blk, 0, stream>>>(proj_w, Wp, ND, ND);
    convert_tr_k<<<dim3(NL * 4 * ND / 32, ND / 32), blk, 0, stream>>>(ff_w1, W1, ND, 4 * ND);
    convert_tr_k<<<dim3(NL * ND / 32, 4 * ND / 32), blk, 0, stream>>>(ff_w2, W2, 4 * ND, ND);
    convert_tr_k<<<dim3(NV / 32, ND / 32), blk, 0, stream>>>(out_w, Wo, ND, NV);

    embed_k<<<NM, blk, 0, stream>>>(x, tok_emb, pos_emb, h);

    for (int l = 0; l < NL; ++l) {
        const size_t wl = (size_t)l * ND * ND;         // 1024 rows x 1024 k
        layernorm_k<<<NM, blk, 0, stream>>>(h, ln1_g + l * ND, ln1_b + l * ND, xn);
        gemm_wmma<0><<<gD, blk, 0, stream>>>(xn, Wq + wl, q, nullptr, nullptr, nullptr,
                                             NM, ND, ND);
        gemm_wmma<0><<<gD, blk, 0, stream>>>(xn, Wk + wl, k, nullptr, nullptr, nullptr,
                                             NM, ND, ND);
        gemm_wmma<4><<<gD, blk, 0, stream>>>(xn, Wv + wl, vT, nullptr, nullptr, nullptr,
                                             NM, ND, ND);
        attention_k<<<gA, blk, 0, stream>>>(q, k, vT, a_attn1 + l * ND, b_attn1 + l * ND, ao);
        gemm_wmma<2><<<gD, blk, 0, stream>>>(ao, Wp + wl, h, proj_b + l * ND,
                                             a_attn2 + l * ND, b_attn2 + l * ND, NM, ND, ND);
        layernorm_k<<<NM, blk, 0, stream>>>(h, ln2_g + l * ND, ln2_b + l * ND, xn);
        gemm_wmma<1><<<gF, blk, 0, stream>>>(xn, W1 + (size_t)l * 4 * ND * ND, f1,
                                             ff_b1 + l * 4 * ND, nullptr, nullptr,
                                             NM, 4 * ND, ND);
        gemm_wmma<2><<<gD, blk, 0, stream>>>(f1, W2 + (size_t)l * ND * 4 * ND, h,
                                             ff_b2 + l * ND, a_ff + l * ND, b_ff + l * ND,
                                             NM, ND, 4 * ND);
    }

    layernorm_k<<<NM, blk, 0, stream>>>(h, lnf_g, lnf_b, xn);
    gemm_wmma<3><<<gV, blk, 0, stream>>>(xn, Wo, d_out, out_b, nullptr, nullptr,
                                         NM, NV, ND);
}
